// MoELayer_54795192762999
// MI455X (gfx1250) — compile-verified, matrix operand order checked by reading, not run on hardware
//
#include <hip/hip_runtime.h>
#include <hip/hip_bf16.h>

// ---------------- problem constants ----------------
#define Bc   16
#define Cc   192
#define Hc   28
#define Wc   28
#define HWc  784            // 28*28
#define Nc   12544          // B*H*W tokens
#define Ec   4
#define HIDc 1152
#define OUTc 192
#define RDc  48

typedef __attribute__((ext_vector_type(16))) _Float16 v16h;
typedef __attribute__((ext_vector_type(8)))  _Float16 v8h;
typedef __attribute__((ext_vector_type(8)))  float    v8f;

// ---------------- workspace layout (bytes) ----------------
#define XS_OFF    ((size_t)0)                          // xs f32 [B][C][HW]      9,633,792
#define XF16_OFF  ((size_t)9633792)                    // x_f16  [N][C]          4,816,896
#define W1T_OFF   ((size_t)14450688)                   // w1t f16 [E][HID][C]    1,769,472
#define W2T_OFF   ((size_t)16220160)                   // w2t f16 [E][OUT][HID]  1,769,472
#define POOL_OFF  ((size_t)17989632)                   // pool f32 [B][C]        12,288
#define SESC_OFF  ((size_t)18001920)                   // sescale f32 [B][C]     12,288
#define TOPP_OFF  ((size_t)18014208)                   // top_p f32 [N]          50,176
#define TOPI_OFF  ((size_t)18064384)                   // top_i i32 [N]          50,176
#define BN1S_OFF  ((size_t)18114560)                   // 192 f32
#define BN1B_OFF  ((size_t)18115328)
#define BN2S_OFF  ((size_t)18116096)
#define BN2B_OFF  ((size_t)18116864)
#define ACC_OFF   ((size_t)18117632)                   // 8 f32: mp[4], mm[4]

// ---------------- LDS layout for k_moe (half-element indices / byte offsets) ----
#define A16_H   0                         // [64][192]  f16  24,576 B
#define H16_H   12288                     // [64][1152] f16 147,456 B
#define BBUF_H  86016                     // [4 waves][2 bufs][16][192] f16 49,152 B
#define A16_B   0
#define BBUF_B  172032
#define MOE_SMEM 221184

__device__ __forceinline__ v16h load_frag16(const _Float16* p0, const _Float16* p1) {
  v8h a = *(const v8h*)p0;
  v8h b = *(const v8h*)p1;
  v16h r;
#pragma unroll
  for (int i = 0; i < 8; ++i) { r[i] = a[i]; r[i + 8] = b[i]; }
  return r;
}

__device__ __forceinline__ float silu_f(float v) { return v / (1.f + __expf(-v)); }

// Async DMA: copy 6144 contiguous bytes global -> LDS (12 x b128 per lane, wave32).
// Tracked by ASYNCcnt; completion is in-order.
__device__ __forceinline__ void async_stage6k(uint32_t lds_byte_off, const _Float16* gsrc, int lane) {
  uint32_t l = lds_byte_off + lane * 16;
  const char* g = (const char*)gsrc + lane * 16;
#pragma unroll
  for (int i = 0; i < 12; ++i) {
    asm volatile("global_load_async_to_lds_b128 %0, %1, off"
                 :: "v"(l), "v"(g) : "memory");
    l += 512; g += 512;
  }
}
#define WAIT_ASYNC_ALL()   asm volatile("s_wait_asynccnt 0x0" ::: "memory")
#define WAIT_ASYNC_12()    asm volatile("s_wait_asynccnt 0xc" ::: "memory")

// ---------------- 0) BN affine precompute + acc zero ----------------
__global__ __launch_bounds__(256) void k_setup(
    const float* g1, const float* b1, const float* rm1, const float* rv1,
    const float* g2, const float* b2, const float* rm2, const float* rv2,
    float* bn1s, float* bn1b, float* bn2s, float* bn2b, float* accs) {
  int t = threadIdx.x;
  if (t < Cc) {
    float s = g1[t] * rsqrtf(rv1[t] + 1e-3f);
    bn1s[t] = s; bn1b[t] = b1[t] - rm1[t] * s;
    float s2 = g2[t] * rsqrtf(rv2[t] + 1e-3f);
    bn2s[t] = s2; bn2b[t] = b2[t] - rm2[t] * s2;
  }
  if (t < 8) accs[t] = 0.f;
}

// ---------------- 1) weight convert + transpose to f16 ----------------
__global__ __launch_bounds__(256) void k_wt(const float* W1, const float* W2,
                                            _Float16* w1t, _Float16* w2t) {
  int i = blockIdx.x * 256 + threadIdx.x;   // 0 .. E*C*HID-1
  if (i >= Ec * Cc * HIDc) return;
  // w1t[e][h][c] = W1[e][c][h]
  int c = i % Cc; int r = i / Cc; int h = r % HIDc; int e = r / HIDc;
  w1t[i] = (_Float16)W1[((size_t)e * Cc + c) * HIDc + h];
  // w2t[e][o][h] = W2[e][h][o]
  int h2 = i % HIDc; int r2 = i / HIDc; int o = r2 % OUTc; int e2 = r2 / OUTc;
  w2t[i] = (_Float16)W2[((size_t)e2 * HIDc + h2) * OUTc + o];
}

// ---------------- 2) depthwise conv + BN1 + SiLU + pooled mean ----------------
__global__ __launch_bounds__(256) void k_dw(const float* __restrict__ x,
                                            const float* __restrict__ dww,
                                            const float* __restrict__ bn1s,
                                            const float* __restrict__ bn1b,
                                            float* __restrict__ xs,
                                            float* __restrict__ pool) {
  int bc = blockIdx.x;            // b*C + c
  int c = bc % Cc;
  __shared__ float wsm[25];
  __shared__ float red[256];
  if (threadIdx.x < 25) wsm[threadIdx.x] = dww[c * 25 + threadIdx.x];
  __syncthreads();
  const float* xp = x + (size_t)bc * HWc;
  float sc = bn1s[c], bi = bn1b[c];
  float accs = 0.f;
  for (int p = threadIdx.x; p < HWc; p += 256) {
    int py = p / Wc, px = p % Wc;
    float a = 0.f;
#pragma unroll
    for (int ky = 0; ky < 5; ++ky) {
      int yy = py + ky - 2;
      if ((unsigned)yy < (unsigned)Hc) {
#pragma unroll
        for (int kx = 0; kx < 5; ++kx) {
          int xx = px + kx - 2;
          if ((unsigned)xx < (unsigned)Wc) a += xp[yy * Wc + xx] * wsm[ky * 5 + kx];
        }
      }
    }
    a = silu_f(a * sc + bi);
    xs[(size_t)bc * HWc + p] = a;
    accs += a;
  }
  red[threadIdx.x] = accs; __syncthreads();
  for (int o = 128; o > 0; o >>= 1) {
    if (threadIdx.x < o) red[threadIdx.x] += red[threadIdx.x + o];
    __syncthreads();
  }
  if (threadIdx.x == 0) pool[bc] = red[0] * (1.f / (float)HWc);
}

// ---------------- 3) SE MLP per batch ----------------
__global__ __launch_bounds__(192) void k_se(const float* __restrict__ pool,
                                            const float* __restrict__ w1, const float* __restrict__ b1,
                                            const float* __restrict__ w2, const float* __restrict__ b2,
                                            float* __restrict__ sescale) {
  int b = blockIdx.x;
  __shared__ float p[Cc];
  __shared__ float s1[RDc];
  int t = threadIdx.x;
  p[t] = pool[b * Cc + t];
  __syncthreads();
  if (t < RDc) {
    float a = b1[t];
    for (int c = 0; c < Cc; ++c) a += p[c] * w1[t * Cc + c];
    s1[t] = silu_f(a);
  }
  __syncthreads();
  float a = b2[t];
  for (int r = 0; r < RDc; ++r) a += s1[r] * w2[t * RDc + r];
  sescale[b * Cc + t] = 1.f / (1.f + __expf(-a));
}

// ---------------- 4) SE-apply + router softmax/argmax + f16 token rows ----------------
__global__ __launch_bounds__(256) void k_router(const float* __restrict__ xs,
                                                const float* __restrict__ sescale,
                                                const float* __restrict__ rw,
                                                const float* __restrict__ rb,
                                                _Float16* __restrict__ xf16,
                                                float* __restrict__ topp, int* __restrict__ topi,
                                                float* __restrict__ accs) {
  int token = blockIdx.x * 8 + (threadIdx.x >> 5);
  int lane = threadIdx.x & 31;
  if (token >= Nc) return;
  int b = token / HWc, hw = token % HWc;
  const float* xp = xs + (size_t)b * Cc * HWc + hw;
  const float* ssp = sescale + b * Cc;
  float l0 = 0.f, l1 = 0.f, l2 = 0.f, l3 = 0.f;
#pragma unroll
  for (int i = 0; i < 6; ++i) {
    int c = lane * 6 + i;
    float v = xp[(size_t)c * HWc] * ssp[c];
    xf16[(size_t)token * Cc + c] = (_Float16)v;
    l0 += v * rw[0 * Cc + c]; l1 += v * rw[1 * Cc + c];
    l2 += v * rw[2 * Cc + c]; l3 += v * rw[3 * Cc + c];
  }
#pragma unroll
  for (int o = 16; o > 0; o >>= 1) {
    l0 += __shfl_xor(l0, o, 32); l1 += __shfl_xor(l1, o, 32);
    l2 += __shfl_xor(l2, o, 32); l3 += __shfl_xor(l3, o, 32);
  }
  if (lane == 0) {
    float lg[4] = { l0 + rb[0], l1 + rb[1], l2 + rb[2], l3 + rb[3] };
    float mx = lg[0];
#pragma unroll
    for (int e = 1; e < 4; ++e) mx = fmaxf(mx, lg[e]);
    float ex[4]; float s = 0.f;
#pragma unroll
    for (int e = 0; e < 4; ++e) { ex[e] = __expf(lg[e] - mx); s += ex[e]; }
    float inv = 1.f / s;
    int best = 0; float bp = ex[0] * inv;
#pragma unroll
    for (int e = 0; e < 4; ++e) {
      float pe = ex[e] * inv;
      atomicAdd(&accs[e], pe);
      if (pe > bp) { bp = pe; best = e; }
    }
    topp[token] = bp;
    topi[token] = best;
    atomicAdd(&accs[4 + best], 1.f);
  }
}

// ---------------- 5) fused expert MLP (M-tile = 64) ----------------
// grid = (196 token-blocks of 64, 4 experts); block = 128 (4 waves)
// GEMM1: async-LDS double-buffered B, 4x register reuse of B across M-tiles
// LN+SiLU in-place on f16 h in LDS; GEMM2: register double-buffered B
__global__ __launch_bounds__(128) void k_moe(const _Float16* __restrict__ xf16,
                                             const _Float16* __restrict__ w1t,
                                             const _Float16* __restrict__ w2t,
                                             const float* __restrict__ lng,
                                             const float* __restrict__ lnb,
                                             const float* __restrict__ topp,
                                             const int* __restrict__ topi,
                                             const float* __restrict__ bn2s,
                                             const float* __restrict__ bn2b,
                                             const float* __restrict__ x_in,
                                             float* __restrict__ out) {
  extern __shared__ char smem[];
  _Float16* sh  = (_Float16*)smem;
  _Float16* a16 = sh + A16_H;           // [64][192]
  _Float16* h16 = sh + H16_H;           // [64][1152]

  const int blk  = blockIdx.x;          // 64-token block
  const int e    = blockIdx.y;          // expert
  const int wave = threadIdx.x >> 5;
  const int lane = threadIdx.x & 31;
  const int lh   = lane >> 4;           // half select
  const int lr   = lane & 15;           // row (M for A, N-row for B, N for C/D)

  // ---- prologue: async-stage A tile (this wave's 16 rows) + first B tile ----
  async_stage6k(A16_B + wave * 6144,
                xf16 + ((size_t)blk * 64 + wave * 16) * Cc, lane);
  async_stage6k(BBUF_B + wave * 12288,
                w1t + ((size_t)e * HIDc + (wave * 18) * 16) * Cc, lane);
  WAIT_ASYNC_ALL();
  __syncthreads();

  // ---- GEMM1: [64 x 192] x [192 x 1152] ----
  for (int j = 0; j < 18; ++j) {
    int nt = wave * 18 + j;             // N-tile 0..71
    if (j < 17) {
      async_stage6k(BBUF_B + wave * 12288 + ((j + 1) & 1) * 6144,
                    w1t + ((size_t)e * HIDc + (nt + 1) * 16) * Cc, lane);
      WAIT_ASYNC_12();                  // tile j complete, tile j+1 in flight
    } else {
      WAIT_ASYNC_ALL();
    }
    const _Float16* bb = sh + BBUF_H + wave * 6144 + (j & 1) * 3072;  // half idx
    v8f acc[4] = {};
#pragma unroll
    for (int ks = 0; ks < 6; ++ks) {
      int k0 = ks * 32 + lh * 8;
      const _Float16* br = bb + lr * Cc;
      v16h bf = load_frag16(br + k0, br + k0 + 16);
#pragma unroll
      for (int mt = 0; mt < 4; ++mt) {
        const _Float16* ar = a16 + (mt * 16 + lr) * Cc;
        v16h af = load_frag16(ar + k0, ar + k0 + 16);
        acc[mt] = __builtin_amdgcn_wmma_f32_16x16x32_f16(
            false, af, false, bf, (short)0, acc[mt], false, false);
      }
    }
#pragma unroll
    for (int mt = 0; mt < 4; ++mt) {
#pragma unroll
      for (int v = 0; v < 8; ++v) {
        int m = v + lh * 8;
        h16[(mt * 16 + m) * HIDc + nt * 16 + lr] = (_Float16)acc[mt][v];
      }
    }
  }
  __syncthreads();

  // ---- LayerNorm(HID) + SiLU, in place on f16 ----
  for (int mm = 0; mm < 16; ++mm) {
    int m = wave * 16 + mm;
    float s = 0.f, ss = 0.f;
    for (int c = lane; c < HIDc; c += 32) {
      float v = (float)h16[m * HIDc + c];
      s += v; ss += v * v;
    }
#pragma unroll
    for (int o = 16; o > 0; o >>= 1) { s += __shfl_xor(s, o, 32); ss += __shfl_xor(ss, o, 32); }
    float mu = s * (1.f / (float)HIDc);
    float var = ss * (1.f / (float)HIDc) - mu * mu;
    float inv = rsqrtf(var + 1e-5f);
    for (int c = lane; c < HIDc; c += 32) {
      float v = ((float)h16[m * HIDc + c] - mu) * inv * lng[e * HIDc + c] + lnb[e * HIDc + c];
      h16[m * HIDc + c] = (_Float16)silu_f(v);
    }
  }
  __syncthreads();

  // ---- GEMM2: [64 x 1152] x [1152 x 192] ----
  for (int jj = 0; jj < 3; ++jj) {
    int nt = wave * 3 + jj;             // N-tile 0..11
    const _Float16* brow = w2t + ((size_t)e * OUTc + nt * 16 + lr) * HIDc;
    if (jj < 2) __builtin_prefetch(brow + 16 * HIDc, 0, 1);
    v8f acc[4] = {};
    v16h bf_cur = load_frag16(brow + lh * 8, brow + lh * 8 + 16);   // ks = 0
    for (int ks = 0; ks < 36; ++ks) {
      v16h bf = bf_cur;
      if (ks < 35) {                     // register double buffer: load next B early
        int k0n = (ks + 1) * 32 + lh * 8;
        bf_cur = load_frag16(brow + k0n, brow + k0n + 16);
      }
      int k0 = ks * 32 + lh * 8;
#pragma unroll
      for (int mt = 0; mt < 4; ++mt) {
        const _Float16* ar = h16 + (size_t)(mt * 16 + lr) * HIDc;
        v16h af = load_frag16(ar + k0, ar + k0 + 16);
        acc[mt] = __builtin_amdgcn_wmma_f32_16x16x32_f16(
            false, af, false, bf, (short)0, acc[mt], false, false);
      }
    }
    int ch = nt * 16 + lr;              // output channel for this lane
    float sc = bn2s[ch], bi = bn2b[ch];
#pragma unroll
    for (int mt = 0; mt < 4; ++mt) {
#pragma unroll
      for (int v = 0; v < 8; ++v) {
        int m = v + lh * 8;
        int token = blk * 64 + mt * 16 + m;
        if (topi[token] == e) {
          int b = token / HWc, hw = token % HWc;
          size_t oidx = ((size_t)b * OUTc + ch) * HWc + hw;
          out[oidx] = x_in[oidx] + acc[mt][v] * topp[token] * sc + bi;
        }
      }
    }
  }
}

// ---------------- 6) aux loss ----------------
__global__ void k_aux(const float* accs, float* out_scalar) {
  float a = 0.f;
#pragma unroll
  for (int e = 0; e < 4; ++e) a += (accs[e] / (float)Nc) * (accs[4 + e] / (float)Nc);
  *out_scalar = 4.f * a;
}

// ---------------- launch ----------------
extern "C" void kernel_launch(void* const* d_in, const int* in_sizes, int n_in,
                              void* d_out, int out_size, void* d_ws, size_t ws_size,
                              hipStream_t stream) {
  const float* x        = (const float*)d_in[0];
  const float* dw_w     = (const float*)d_in[1];
  const float* bn1_g    = (const float*)d_in[2];
  const float* bn1_b    = (const float*)d_in[3];
  const float* bn1_rm   = (const float*)d_in[4];
  const float* bn1_rv   = (const float*)d_in[5];
  const float* se_w1    = (const float*)d_in[6];
  const float* se_b1    = (const float*)d_in[7];
  const float* se_w2    = (const float*)d_in[8];
  const float* se_b2    = (const float*)d_in[9];
  const float* router_w = (const float*)d_in[10];
  const float* router_b = (const float*)d_in[11];
  const float* W1       = (const float*)d_in[12];
  const float* ln_g     = (const float*)d_in[13];
  const float* ln_b     = (const float*)d_in[14];
  const float* W2       = (const float*)d_in[15];
  const float* bn2_g    = (const float*)d_in[16];
  const float* bn2_b    = (const float*)d_in[17];
  const float* bn2_rm   = (const float*)d_in[18];
  const float* bn2_rv   = (const float*)d_in[19];

  char* ws = (char*)d_ws;
  float*    xs      = (float*)(ws + XS_OFF);
  _Float16* xf16    = (_Float16*)(ws + XF16_OFF);
  _Float16* w1t     = (_Float16*)(ws + W1T_OFF);
  _Float16* w2t     = (_Float16*)(ws + W2T_OFF);
  float*    pool    = (float*)(ws + POOL_OFF);
  float*    sescale = (float*)(ws + SESC_OFF);
  float*    topp    = (float*)(ws + TOPP_OFF);
  int*      topi    = (int*)(ws + TOPI_OFF);
  float*    bn1s    = (float*)(ws + BN1S_OFF);
  float*    bn1b    = (float*)(ws + BN1B_OFF);
  float*    bn2s    = (float*)(ws + BN2S_OFF);
  float*    bn2b    = (float*)(ws + BN2B_OFF);
  float*    accs    = (float*)(ws + ACC_OFF);

  float* out = (float*)d_out;

  k_setup<<<1, 256, 0, stream>>>(bn1_g, bn1_b, bn1_rm, bn1_rv,
                                 bn2_g, bn2_b, bn2_rm, bn2_rv,
                                 bn1s, bn1b, bn2s, bn2b, accs);
  k_wt<<<(Ec * Cc * HIDc + 255) / 256, 256, 0, stream>>>(W1, W2, w1t, w2t);
  k_dw<<<Bc * Cc, 256, 0, stream>>>(x, dw_w, bn1s, bn1b, xs, pool);
  k_se<<<Bc, 192, 0, stream>>>(pool, se_w1, se_b1, se_w2, se_b2, sescale);
  k_router<<<Nc / 8, 256, 0, stream>>>(xs, sescale, router_w, router_b,
                                       xf16, topp, topi, accs);
  k_moe<<<dim3(Nc / 64, Ec), 128, MOE_SMEM, stream>>>(xf16, w1t, w2t, ln_g, ln_b,
                                                      topp, topi, bn2s, bn2b, x, out);
  k_aux<<<1, 1, 0, stream>>>(accs, out + (size_t)Bc * OUTc * HWc);
}
